// MentionScore_19430432047649
// MI455X (gfx1250) — matrix operand order repaired and checked by reading, not compiled, not run on hardware
//
#include <hip/hip_runtime.h>

typedef __attribute__((ext_vector_type(16))) __bf16 v16bf;
typedef __attribute__((ext_vector_type(8)))  float  v8f;

#define T_TOK   2048
#define LMAX    10
#define NSTART  (T_TOK - LMAX + 1)             // 2039
#define NSPANS  (NSTART * LMAX)                // 20390
#define NPAD    20480                          // NSPANS padded to multiple of 128
#define HID     150
#define HIDP    160                            // padded hidden dim (10 WMMA col tiles)
#define ADIM    1024
#define DDIM    20

__device__ __forceinline__ unsigned short f32_to_bf16_bits(float f) {
  unsigned int u = __builtin_bit_cast(unsigned int, f);
  unsigned int r = u + 0x7FFFu + ((u >> 16) & 1u);   // round-to-nearest-even
  return (unsigned short)(r >> 16);
}

// ---------------------------------------------------------------------------
// Weight prep: f32 row-major [K x N(<=150)] -> bf16 swizzled [160][Kp],
// dst[col*Kp + k] = bf16(W[k*N + col]), zero padded.
// ---------------------------------------------------------------------------
__global__ void weight_swizzle(const float* __restrict__ W, int K, int N,
                               unsigned short* __restrict__ dst, int Kp)
{
  int idx = blockIdx.x * blockDim.x + threadIdx.x;
  if (idx >= HIDP * Kp) return;
  int col = idx / Kp, k = idx % Kp;
  float v = (col < N && k < K) ? W[(long)k * N + col] : 0.f;
  dst[idx] = f32_to_bf16_bits(v);
}

// ---------------------------------------------------------------------------
// Shared WMMA mainloop: 256 threads / 8 waves, 128x160 C tile, A lda == K.
// ---------------------------------------------------------------------------
template<int K>
__device__ __forceinline__ void wmma_mainloop(const float* __restrict__ A,
                                              const unsigned short* __restrict__ Bsw,
                                              long row0, int tid,
                                              unsigned short* As, unsigned short* Bs,
                                              v8f (&acc)[10])
{
  const int lane   = tid & 31;
  const int wave   = tid >> 5;
  const int half16 = lane >> 4;
  const int l16    = lane & 15;

  const float* aptr = A + (row0 + (tid >> 3)) * K + (tid & 7) * 4;
  const unsigned short* bsrc0 = Bsw + (long)(tid >> 2) * K + (tid & 3) * 8;
  const unsigned short* bsrc1 = Bsw + (long)((tid + 256) >> 2) * K + ((tid + 256) & 3) * 8;
  const unsigned short* bsrc2 = Bsw + (long)((tid + 512) >> 2) * K + ((tid + 512) & 3) * 8;
  const int bdst0 = (tid >> 2) * 32 + (tid & 3) * 8;
  const int bdst1 = ((tid + 256) >> 2) * 32 + ((tid + 256) & 3) * 8;
  const int bdst2 = ((tid + 512) >> 2) * 32 + ((tid + 512) & 3) * 8;
  const bool b2ok = (tid < 128);
  const int abase = (wave * 16 + l16) * 32 + half16 * 8;

  for (int k0 = 0; k0 < K; k0 += 32) {
    __syncthreads();
#pragma unroll
    for (int i = 0; i < 4; ++i) {
      const float4 v = *(const float4*)(aptr + (long)i * 32 * K + k0);
      unsigned int lo = (unsigned int)f32_to_bf16_bits(v.x) |
                        ((unsigned int)f32_to_bf16_bits(v.y) << 16);
      unsigned int hi = (unsigned int)f32_to_bf16_bits(v.z) |
                        ((unsigned int)f32_to_bf16_bits(v.w) << 16);
      *(uint2*)&As[((tid >> 3) + 32 * i) * 32 + (tid & 7) * 4] = make_uint2(lo, hi);
    }
    *(uint4*)&Bs[bdst0] = *(const uint4*)(bsrc0 + k0);
    *(uint4*)&Bs[bdst1] = *(const uint4*)(bsrc1 + k0);
    if (b2ok) *(uint4*)&Bs[bdst2] = *(const uint4*)(bsrc2 + k0);
    __syncthreads();

    union { v16bf v; uint4 q[2]; } ua;
    ua.q[0] = *(const uint4*)&As[abase];
    ua.q[1] = *(const uint4*)&As[abase + 16];
#pragma unroll
    for (int t = 0; t < 10; ++t) {
      union { v16bf v; uint4 q[2]; } ub;
      const int bbase = (t * 16 + l16) * 32 + half16 * 8;
      ub.q[0] = *(const uint4*)&Bs[bbase];
      ub.q[1] = *(const uint4*)&Bs[bbase + 16];
      acc[t] = __builtin_amdgcn_wmma_f32_16x16x32_bf16(
          false, ua.v, false, ub.v, (short)0, acc[t], false, false);
    }
  }
}

// C[M x 160] = act(A @ Bsw + bias) -> f32, ld 160.
template<int K>
__global__ __launch_bounds__(256)
void gemm_bf16_wmma(const float* __restrict__ A,
                    const unsigned short* __restrict__ Bsw,
                    const float* __restrict__ bias, int nbias,
                    float* __restrict__ C, int relu)
{
  __shared__ unsigned short As[128 * 32];
  __shared__ unsigned short Bs[160 * 32];
  const int tid    = threadIdx.x;
  const int wave   = tid >> 5;
  const int half16 = (tid & 31) >> 4;
  const int l16    = tid & 15;
  const long row0  = (long)blockIdx.x * 128;

  v8f acc[10] = {};
  wmma_mainloop<K>(A, Bsw, row0, tid, As, Bs, acc);

#pragma unroll
  for (int t = 0; t < 10; ++t) {
    int col = t * 16 + l16;
    float bv = (bias != nullptr && col < nbias) ? bias[col] : 0.f;
#pragma unroll
    for (int vv = 0; vv < 8; ++vv) {
      long gr = row0 + wave * 16 + vv + half16 * 8;
      float val = acc[t][vv] + bv;
      if (relu) val = fmaxf(val, 0.f);
      C[gr * HIDP + col] = val;
    }
  }
}

// Fused layers 2+3: out[row] = relu(A @ Bsw + b2) . w3 + b3  (row < Mvalid)
template<int K>
__global__ __launch_bounds__(256)
void gemm_bf16_wmma_dot(const float* __restrict__ A,
                        const unsigned short* __restrict__ Bsw,
                        const float* __restrict__ bias2,
                        const float* __restrict__ w3,
                        const float* __restrict__ bias3,
                        float* __restrict__ out, int Mvalid)
{
  __shared__ unsigned short As[128 * 32];
  __shared__ unsigned short Bs[160 * 32];
  const int tid    = threadIdx.x;
  const int wave   = tid >> 5;
  const int half16 = (tid & 31) >> 4;
  const int l16    = tid & 15;
  const long row0  = (long)blockIdx.x * 128;

  v8f acc[10] = {};
  wmma_mainloop<K>(A, Bsw, row0, tid, As, Bs, acc);

  float partial[8] = {};
#pragma unroll
  for (int t = 0; t < 10; ++t) {
    int col = t * 16 + l16;
    float b2v = (col < HID) ? bias2[col] : 0.f;
    float w3v = (col < HID) ? w3[col] : 0.f;
#pragma unroll
    for (int vv = 0; vv < 8; ++vv)
      partial[vv] += fmaxf(acc[t][vv] + b2v, 0.f) * w3v;
  }
#pragma unroll
  for (int vv = 0; vv < 8; ++vv) {
#pragma unroll
    for (int off = 1; off < 16; off <<= 1)
      partial[vv] += __shfl_xor(partial[vv], off, 32);
  }
  if (l16 == 0) {
    float b3v = bias3[0];
#pragma unroll
    for (int vv = 0; vv < 8; ++vv) {
      long row = row0 + wave * 16 + vv + half16 * 8;
      if (row < Mvalid) out[row] = partial[vv] + b3v;
    }
  }
}

// widthW[r][j] = b1[j] + sum_d width_table[r][d] * W1d[d][j]   (9 x HIDP)
__global__ void width_proj(const float* __restrict__ table,
                           const float* __restrict__ W1d,
                           const float* __restrict__ b1,
                           float* __restrict__ widthW)
{
  int idx = blockIdx.x * blockDim.x + threadIdx.x;
  if (idx >= 9 * HIDP) return;
  int r = idx / HIDP, j = idx % HIDP;
  float s = 0.f;
  if (j < HID) {
    s = b1[j];
#pragma unroll
    for (int d = 0; d < DDIM; ++d)
      s += table[r * DDIM + d] * W1d[d * HID + j];
  }
  widthW[idx] = s;
}

// One block per start position; produces all LMAX widths with incremental
// softmax pooling (single shared max over the 10-token window; softmax over
// the first w tokens = e_l / prefix-sum, identical to per-width renorm).
__global__ void span_h1(const float* __restrict__ attns,
                        const float* __restrict__ S1,
                        const float* __restrict__ S2,
                        const float* __restrict__ P,
                        const float* __restrict__ widthW,
                        float* __restrict__ H1)
{
  const int start = blockIdx.x;          // 0..NSTART-1
  const int j     = threadIdx.x;         // 0..159

  float a[LMAX];
#pragma unroll
  for (int l = 0; l < LMAX; ++l) a[l] = attns[start + l];
  float m = a[0];
#pragma unroll
  for (int l = 1; l < LMAX; ++l) m = fmaxf(m, a[l]);
  float e[LMAX];
#pragma unroll
  for (int l = 0; l < LMAX; ++l) e[l] = __expf(a[l] - m);

  const int binofw[LMAX] = {1, 2, 3, 4, 4, 4, 4, 5, 5, 5};  // bucket(w), w=1..10

  const float s1 = (j < HID) ? S1[(long)start * HIDP + j] : 0.f;
  float pool = 0.f, csum = 0.f;
  const long outbase = (long)start * LMAX * HIDP + j;
#pragma unroll
  for (int w = 1; w <= LMAX; ++w) {
    const int idx = start + w - 1;
    csum += e[w - 1];
    float h = 0.f;
    if (j < HID) {
      pool += e[w - 1] * P[(long)idx * HIDP + j];
      h = s1 + S2[(long)idx * HIDP + j] + pool / csum
        + widthW[binofw[w - 1] * HIDP + j];
      h = fmaxf(h, 0.f);
    }
    H1[outbase + (long)(w - 1) * HIDP] = h;
  }
}

// Zero H1 pad rows [NSPANS, NPAD) so guard-free GEMM reads zeros.
__global__ void h1_zero_pad(float* __restrict__ H1)
{
  int idx = blockIdx.x * blockDim.x + threadIdx.x;
  if (idx < (NPAD - NSPANS) * HIDP)
    H1[(long)NSPANS * HIDP + idx] = 0.f;
}

extern "C" void kernel_launch(void* const* d_in, const int* in_sizes, int n_in,
                              void* d_out, int out_size, void* d_ws, size_t ws_size,
                              hipStream_t stream)
{
  const float* states      = (const float*)d_in[0];
  const float* embeds      = (const float*)d_in[1];
  const float* attn_W1     = (const float*)d_in[2];
  const float* attn_b1     = (const float*)d_in[3];
  const float* attn_W2     = (const float*)d_in[4];
  const float* attn_b2     = (const float*)d_in[5];
  const float* attn_W3     = (const float*)d_in[6];
  const float* attn_b3     = (const float*)d_in[7];
  const float* width_table = (const float*)d_in[8];
  const float* score_W1    = (const float*)d_in[9];
  const float* score_b1    = (const float*)d_in[10];
  const float* score_W2    = (const float*)d_in[11];
  const float* score_b2    = (const float*)d_in[12];
  const float* score_W3    = (const float*)d_in[13];
  const float* score_b3    = (const float*)d_in[14];

  // f32 scratch
  float* ws = (float*)d_ws;
  float* HA1    = ws; ws += (size_t)T_TOK * HIDP;
  float* attns  = ws; ws += T_TOK;
  float* S1     = ws; ws += (size_t)T_TOK * HIDP;
  float* S2     = ws; ws += (size_t)T_TOK * HIDP;
  float* P      = ws; ws += (size_t)T_TOK * HIDP;
  float* widthW = ws; ws += 9 * HIDP;
  float* H1     = ws; ws += (size_t)NPAD * HIDP;
  // bf16 swizzled weights
  unsigned short* us = (unsigned short*)ws;
  unsigned short* W1sw_attn = us; us += (size_t)HIDP * ADIM;
  unsigned short* W2sw_attn = us; us += (size_t)HIDP * HIDP;
  unsigned short* sW1a      = us; us += (size_t)HIDP * ADIM;
  unsigned short* sW1b      = us; us += (size_t)HIDP * ADIM;
  unsigned short* sW1c      = us; us += (size_t)HIDP * ADIM;
  unsigned short* sW2       = us; us += (size_t)HIDP * HIDP;

  const int swz_big   = (HIDP * ADIM + 255) / 256;
  const int swz_small = (HIDP * HIDP + 255) / 256;

  // ---- weight prep (bf16 swizzle, zero-padded) ----
  weight_swizzle<<<swz_big,   256, 0, stream>>>(attn_W1, ADIM, HID, W1sw_attn, ADIM);
  weight_swizzle<<<swz_small, 256, 0, stream>>>(attn_W2, HID,  HID, W2sw_attn, HIDP);
  weight_swizzle<<<swz_big,   256, 0, stream>>>(score_W1,                      ADIM, HID, sW1a, ADIM);
  weight_swizzle<<<swz_big,   256, 0, stream>>>(score_W1 + (size_t)1024 * HID, ADIM, HID, sW1b, ADIM);
  weight_swizzle<<<swz_big,   256, 0, stream>>>(score_W1 + (size_t)2048 * HID, ADIM, HID, sW1c, ADIM);
  weight_swizzle<<<swz_small, 256, 0, stream>>>(score_W2, HID, HID, sW2, HIDP);
  width_proj<<<(9 * HIDP + 255) / 256, 256, 0, stream>>>(
      width_table, score_W1 + (size_t)3072 * HID, score_b1, widthW);
  h1_zero_pad<<<((NPAD - NSPANS) * HIDP + 255) / 256, 256, 0, stream>>>(H1);

  const dim3 blk(256);
  const int tokBlocks  = T_TOK / 128;   // 16
  const int spanBlocks = NPAD / 128;    // 160

  // ---- attention MLP: layer 1 GEMM, layers 2+3 fused ----
  gemm_bf16_wmma<ADIM><<<tokBlocks, blk, 0, stream>>>(states, W1sw_attn, attn_b1, HID, HA1, 1);
  gemm_bf16_wmma_dot<HIDP><<<tokBlocks, blk, 0, stream>>>(
      HA1, W2sw_attn, attn_b2, attn_W3, attn_b3, attns, T_TOK);

  // ---- token projections through score_W1 slices ----
  gemm_bf16_wmma<ADIM><<<tokBlocks, blk, 0, stream>>>(states, sW1a, nullptr, 0, S1, 0);
  gemm_bf16_wmma<ADIM><<<tokBlocks, blk, 0, stream>>>(states, sW1b, nullptr, 0, S2, 0);
  gemm_bf16_wmma<ADIM><<<tokBlocks, blk, 0, stream>>>(embeds, sW1c, nullptr, 0, P,  0);

  // ---- per-start pooling + layer-1 combine (all widths per block) ----
  span_h1<<<NSTART, dim3(HIDP), 0, stream>>>(attns, S1, S2, P, widthW, H1);

  // ---- mention MLP layers 2+3 fused ----
  gemm_bf16_wmma_dot<HIDP><<<spanBlocks, blk, 0, stream>>>(
      H1, sW2, score_b2, score_W3, score_b3, (float*)d_out, NSPANS);
}
